// HyperNetworkDecoder_43327630082494
// MI455X (gfx1250) — matrix operand order repaired and checked by reading, not compiled, not run on hardware
//
#include <hip/hip_runtime.h>
#include <hip/hip_bf16.h>
#include <math.h>

// ---------------------------------------------------------------------------
// HyperNetworkDecoder for MI455X (gfx1250, wave32, WMMA + async LDS DMA)
//
//   1) mlp_ln_relu : h1 = relu(LN(z @ W1 + b1))        (128x256)
//   2) mlp_ln_relu : h2 = relu(LN(h1 @ W2 + b2))       (128x256)
//   3) mlp_out     : params = (h2 @ W3 + b3)*scale     (128x2616)
//   4) ca_kernel   : per-batch neural CA, 32 steps fully LDS-resident.
//      Convs as implicit GEMM via v_wmma_f32_16x16x32_bf16, position-major K
//      (K = kk*12+ci) + channel-last LDS images => B-operand quads are single
//      aligned ds_load_b64s. Params in / output out use async LDS DMA
//      (global_load_async_to_lds_b128 / global_store_async_from_lds_b128).
// ---------------------------------------------------------------------------

typedef __attribute__((ext_vector_type(16))) __bf16 v16bf;
typedef __attribute__((ext_vector_type(8)))  float  v8f;

#define CCH   12              // channels
#define HWD   56              // grid H=W
#define PHW   58              // padded H=W
#define NPIX  (HWD*HWD)       // 3136
#define PPIX  (PHW*PHW)       // 3364
#define WS9   108             // C*9 = K per conv
#define WSTOT 1296            // C*C*9
#define PTOT  2616            // 2*WS + 2*C
#define NTILE (NPIX/16)       // 196
#define ROWHW (PHW*CCH)       // 696 halfwords per padded row

#define GFP_BYTES (NPIX*CCH*4)   // 150528  fp32 grid, channel-last [pix][12]
#define PB_BYTES  (PPIX*CCH*2)   // 80736   bf16 padded image, channel-last
#define PRM_BYTES (PTOT*4)       // 10464   staged params row
#define SMEM_BYTES (GFP_BYTES + 2*PB_BYTES + PRM_BYTES) // 322464 <= 320KB

static __device__ __forceinline__ unsigned short f2bfbits(float f) {
  unsigned int u = __builtin_bit_cast(unsigned int, f);
  unsigned int r = (u + 0x7FFFu + ((u >> 16) & 1u)) >> 16;  // RNE
  return (unsigned short)r;
}
static __device__ __forceinline__ __bf16 bfb(float f) {
  return __builtin_bit_cast(__bf16, f2bfbits(f));
}
static __device__ __forceinline__ unsigned int packbf2(float a, float b) {
  return (unsigned int)f2bfbits(a) | ((unsigned int)f2bfbits(b) << 16);
}
static __device__ __forceinline__ unsigned long long packbf4(float a, float b,
                                                            float c, float d) {
  return (unsigned long long)packbf2(a, b) |
         ((unsigned long long)packbf2(c, d) << 32);
}

// LDS byte offset of a pointer that lives in shared memory (addrspace 3).
static __device__ __forceinline__ unsigned lds_offset(const void* p) {
  return (unsigned)(unsigned long long)(const __attribute__((address_space(3))) char*)p;
}

// ---------------------------------------------------------------------------
// B-operand gather: K = kk*12 + ci, image channel-last => each aligned quad
// of 4 K-values is 8 contiguous bytes inside one kh-row (boundaries 36,72 are
// multiples of 4) => one ds_load_b64 per quad, offsets compile-time.
// cb = (y*58 + x) * 12 (halfwords). HI = lane>=16.
// ---------------------------------------------------------------------------
template <int HI>
static __device__ __forceinline__ void gather_b(
    const unsigned short* __restrict__ S, int cb, v16bf bv[4]) {
#pragma unroll
  for (int c = 0; c < 4; ++c) {
    unsigned long long q0 = 0, q1 = 0, q2 = 0, q3 = 0;
#pragma unroll
    for (int qi = 0; qi < 4; ++qi) {
      const int Kg = 16 * HI + 32 * c + 4 * qi;
      if (Kg <= 104) {  // quad fully valid (Kg..Kg+3 < 108)
        const int kh  = Kg / 36;
        const int off = kh * ROWHW + (Kg - 36 * kh);
        const unsigned long long v =
            *(const unsigned long long*)(S + cb + off);
        if (qi == 0) q0 = v; else if (qi == 1) q1 = v;
        else if (qi == 2) q2 = v; else q3 = v;
      }
    }
    struct Q4 { unsigned long long a, b, c2, d; };
    Q4 t{q0, q1, q2, q3};
    bv[c] = __builtin_bit_cast(v16bf, t);
  }
}

// A-operand (generated conv weights) in the contiguous k order gather_b uses:
// lane row M = lane&15 (out channel), element e -> K = e + 16*HI.
// Weight flat index for [co][ci][kh][kw] with K = kk*12+ci.
template <int HI>
static __device__ __forceinline__ void load_a(
    const float* __restrict__ Pw, int co, v16bf a[4]) {
#pragma unroll
  for (int c = 0; c < 4; ++c) {
    v16bf w;
#pragma unroll
    for (int e = 0; e < 16; ++e) {
      const int K = e + 16 * HI + 32 * c;
      float f = 0.0f;
      if (K < WS9) {
        const int ci = K % 12, kk = K / 12;
        if (co < CCH) f = Pw[co * WS9 + ci * 9 + kk];
      }
      w[e] = bfb(f);
    }
    a[c] = w;
  }
}

// ---------------------------------------------------------------------------
// MLP layer with LayerNorm + ReLU. One block per batch row, 256 threads.
// ---------------------------------------------------------------------------
__global__ void __launch_bounds__(256) mlp_ln_relu(
    const float* __restrict__ X, const float* __restrict__ W,
    const float* __restrict__ bias, const float* __restrict__ gam,
    const float* __restrict__ bet, float* __restrict__ Y, int K) {
  const int row = blockIdx.x, tid = threadIdx.x;
  __shared__ float xs[256];
  __shared__ float red[256];
  for (int i = tid; i < K; i += 256) xs[i] = X[row * K + i];
  __syncthreads();
  float o = bias[tid];
  for (int k = 0; k < K; ++k) o = fmaf(xs[k], W[k * 256 + tid], o);
  red[tid] = o; __syncthreads();
  for (int s = 128; s > 0; s >>= 1) { if (tid < s) red[tid] += red[tid + s]; __syncthreads(); }
  const float m = red[0] * (1.0f / 256.0f);
  __syncthreads();
  red[tid] = o * o; __syncthreads();
  for (int s = 128; s > 0; s >>= 1) { if (tid < s) red[tid] += red[tid + s]; __syncthreads(); }
  const float var = red[0] * (1.0f / 256.0f) - m * m;
  const float y = (o - m) * rsqrtf(var + 1e-5f) * gam[tid] + bet[tid];
  Y[row * 256 + tid] = fmaxf(y, 0.0f);
}

__global__ void __launch_bounds__(256) mlp_out(
    const float* __restrict__ H, const float* __restrict__ W3,
    const float* __restrict__ b3, const float* __restrict__ wscale,
    float* __restrict__ P) {
  const int row = blockIdx.x, tid = threadIdx.x;
  __shared__ float hs[256];
  hs[tid] = H[row * 256 + tid];
  __syncthreads();
  const float sc = wscale[0];
  for (int j = tid; j < PTOT; j += 256) {
    float a = b3[j];
    for (int k = 0; k < 256; ++k) a = fmaf(hs[k], W3[k * PTOT + j], a);
    P[row * PTOT + j] = a * sc;
  }
}

// halo cell enumeration: 228 cells = rows {0,57} x 58 cols + cols {0,57} x 56
static __device__ __forceinline__ void halo_cell(int i, int& py, int& px) {
  if (i < 116) { py = (i < 58) ? 0 : 57; px = i % 58; }
  else { const int j = i - 116; px = (j < 56) ? 0 : 57; py = 1 + (j % 56); }
}

// ---------------------------------------------------------------------------
// Neural-CA kernel: one WG (256 thr = 8 waves) per batch element.
// ---------------------------------------------------------------------------
__global__ void __launch_bounds__(256) ca_kernel(
    const float* __restrict__ params, const float* __restrict__ initpat,
    const int* __restrict__ nsteps_p, float* __restrict__ out) {
  const int b    = blockIdx.x;
  const int tid  = threadIdx.x;
  const int lane = tid & 31;
  const int wave = tid >> 5;     // 0..7
  const int hi   = (lane >> 4) & 1;
  const int nlc  = lane & 15;

  extern __shared__ char smem[];
  float*          GFPL = (float*)smem;                                // [3136][12] fp32
  unsigned short* PBL  = (unsigned short*)(smem + GFP_BYTES);         // [58][58][12] bf16
  unsigned short* HBL  = PBL + PPIX * CCH;                            // [58][58][12] bf16
  float*          PRM  = (float*)(smem + GFP_BYTES + 2 * PB_BYTES);   // [2616]

  const float* P = params + b * PTOT;
  __builtin_prefetch(initpat + tid, 0, 1);  // global_prefetch_b8

  // --- async DMA: stage params row (10464 B = 654 x 16B) into LDS ---------
  {
    const unsigned long long gbase = (unsigned long long)P;
    const unsigned prm_lds = lds_offset(PRM);
    for (int i = tid; i < PTOT / 4; i += 256) {  // 654 chunks of b128
      const unsigned dst  = prm_lds + (unsigned)i * 16u;
      const unsigned voff = (unsigned)i * 16u;
      asm volatile("global_load_async_to_lds_b128 %0, %1, %2"
                   :: "v"(dst), "v"(voff), "s"(gbase) : "memory");
    }
  }
  // overlap: zero the grid while the DMA is in flight
  {
    float4 z4 = {0.f, 0.f, 0.f, 0.f};
    for (int i = tid; i < NPIX * 3; i += 256) ((float4*)GFPL)[i] = z4;
  }
  asm volatile("s_wait_asynccnt 0" ::: "memory");
  __syncthreads();

  // 7x7 initial pattern at rows/cols 24..30; weights -> A registers
  for (int i = tid; i < CCH * 49; i += 256) {
    const int ch = i / 49, rr = (i / 7) % 7, cc = i % 7;
    GFPL[((24 + rr) * HWD + (24 + cc)) * CCH + ch] = initpat[i];
  }
  v16bf a1[4], a2[4];
  if (hi == 0) {
    load_a<0>(PRM, nlc, a1);
    load_a<0>(PRM + WSTOT + CCH, nlc, a2);
  } else {
    load_a<1>(PRM, nlc, a1);
    load_a<1>(PRM + WSTOT + CCH, nlc, a2);
  }
  float bias1[8], bias2[8];
#pragma unroll
  for (int r = 0; r < 8; ++r) {
    const int ch = r + 8 * hi;
    bias1[r] = (ch < CCH) ? PRM[WSTOT + ch] : 0.0f;
    bias2[r] = (ch < CCH) ? PRM[2 * WSTOT + CCH + ch] : 0.0f;
  }
  __syncthreads();

  // full padded bf16 copy of grid (wrap), one cell per thread iteration
  for (int cell = tid; cell < PPIX; cell += 256) {
    const int py = cell / PHW, px = cell % PHW;
    const int gy = (py + 55) % 56, gx = (px + 55) % 56;
    const float4* g = (const float4*)(GFPL + (gy * HWD + gx) * CCH);
    const float4 c0 = g[0], c1 = g[1], c2 = g[2];
    unsigned long long* dst = (unsigned long long*)(PBL + cell * CCH);
    dst[0] = packbf4(c0.x, c0.y, c0.z, c0.w);
    dst[1] = packbf4(c1.x, c1.y, c1.z, c1.w);
    dst[2] = packbf4(c2.x, c2.y, c2.z, c2.w);
  }
  __syncthreads();

  const int nsteps = nsteps_p[0];
  for (int s = 0; s < nsteps; ++s) {
    // ---- conv1: PBL -> HBL interior, relu(conv + b1) ----
    for (int t = wave; t < NTILE; t += 8) {
      const int p = t * 16 + nlc;
      const int y = p / HWD, x = p % HWD;
      const int cb = (y * PHW + x) * CCH;
      v16bf bv[4];
      if (hi == 0) gather_b<0>(PBL, cb, bv);
      else         gather_b<1>(PBL, cb, bv);
      v8f acc = {};
#pragma unroll
      for (int c = 0; c < 4; ++c)
        acc = __builtin_amdgcn_wmma_f32_16x16x32_bf16(
            false, a1[c], false, bv[c], (short)0, acc, false, false);
      float fr[8];
#pragma unroll
      for (int r = 0; r < 8; ++r) fr[r] = fmaxf(acc[r] + bias1[r], 0.0f);
      const int ob = ((y + 1) * PHW + (x + 1)) * CCH;
      if (hi == 0) {
        *(unsigned long long*)(HBL + ob)     = packbf4(fr[0], fr[1], fr[2], fr[3]);
        *(unsigned long long*)(HBL + ob + 4) = packbf4(fr[4], fr[5], fr[6], fr[7]);
      } else {
        *(unsigned long long*)(HBL + ob + 8) = packbf4(fr[0], fr[1], fr[2], fr[3]);
      }
    }
    __syncthreads();
    // ---- HBL wrap halo (sources are interior cells) ----
    for (int i = tid; i < 228; i += 256) {
      int py, px; halo_cell(i, py, px);
      const int sy = ((py + 55) % 56) + 1, sx = ((px + 55) % 56) + 1;
      const unsigned long long* src =
          (const unsigned long long*)(HBL + (sy * PHW + sx) * CCH);
      unsigned long long* dst =
          (unsigned long long*)(HBL + (py * PHW + px) * CCH);
      dst[0] = src[0]; dst[1] = src[1]; dst[2] = src[2];
    }
    __syncthreads();
    // ---- conv2: HBL -> grid accumulate (fp32) + refresh PBL interior ----
    for (int t = wave; t < NTILE; t += 8) {
      const int p = t * 16 + nlc;
      const int y = p / HWD, x = p % HWD;
      const int cb = (y * PHW + x) * CCH;
      v16bf bv[4];
      if (hi == 0) gather_b<0>(HBL, cb, bv);
      else         gather_b<1>(HBL, cb, bv);
      v8f acc = {};
#pragma unroll
      for (int c = 0; c < 4; ++c)
        acc = __builtin_amdgcn_wmma_f32_16x16x32_bf16(
            false, a2[c], false, bv[c], (short)0, acc, false, false);
      float4* G = (float4*)(GFPL + (y * HWD + x) * CCH);
      const int pb = ((y + 1) * PHW + (x + 1)) * CCH;
      if (hi == 0) {
        float4 g0 = G[0], g1 = G[1];
        g0.x += acc[0] + bias2[0]; g0.y += acc[1] + bias2[1];
        g0.z += acc[2] + bias2[2]; g0.w += acc[3] + bias2[3];
        g1.x += acc[4] + bias2[4]; g1.y += acc[5] + bias2[5];
        g1.z += acc[6] + bias2[6]; g1.w += acc[7] + bias2[7];
        G[0] = g0; G[1] = g1;
        *(unsigned long long*)(PBL + pb)     = packbf4(g0.x, g0.y, g0.z, g0.w);
        *(unsigned long long*)(PBL + pb + 4) = packbf4(g1.x, g1.y, g1.z, g1.w);
      } else {
        float4 g2 = G[2];
        g2.x += acc[0] + bias2[0]; g2.y += acc[1] + bias2[1];
        g2.z += acc[2] + bias2[2]; g2.w += acc[3] + bias2[3];
        G[2] = g2;
        *(unsigned long long*)(PBL + pb + 8) = packbf4(g2.x, g2.y, g2.z, g2.w);
      }
    }
    __syncthreads();
    // ---- PBL wrap halo from fp32 grid ----
    for (int i = tid; i < 228; i += 256) {
      int py, px; halo_cell(i, py, px);
      const int gy = (py + 55) % 56, gx = (px + 55) % 56;
      const float4* g = (const float4*)(GFPL + (gy * HWD + gx) * CCH);
      const float4 c0 = g[0], c1 = g[1], c2 = g[2];
      unsigned long long* dst =
          (unsigned long long*)(PBL + (py * PHW + px) * CCH);
      dst[0] = packbf4(c0.x, c0.y, c0.z, c0.w);
      dst[1] = packbf4(c1.x, c1.y, c1.z, c1.w);
      dst[2] = packbf4(c2.x, c2.y, c2.z, c2.w);
    }
    __syncthreads();
  }

  // ---- output: sigmoid of channel 0, staged in LDS then async DMA out ----
  float* OutS = (float*)PBL;  // reuse 12544 B of PBL as fp32 staging
  for (int p = tid; p < NPIX; p += 256) {
    const float g = GFPL[p * CCH];
    OutS[p] = 1.0f / (1.0f + expf(-g));
  }
  __syncthreads();
  {
    const unsigned long long gbase = (unsigned long long)(out + b * NPIX);
    const unsigned src_lds = lds_offset(OutS);
    for (int i = tid; i < NPIX / 4; i += 256) {  // 784 chunks of b128
      const unsigned src  = src_lds + (unsigned)i * 16u;
      const unsigned voff = (unsigned)i * 16u;
      asm volatile("global_store_async_from_lds_b128 %0, %1, %2"
                   :: "v"(voff), "v"(src), "s"(gbase) : "memory");
    }
  }
  asm volatile("s_wait_asynccnt 0" ::: "memory");
}

// ---------------------------------------------------------------------------
extern "C" void kernel_launch(void* const* d_in, const int* in_sizes, int n_in,
                              void* d_out, int out_size, void* d_ws, size_t ws_size,
                              hipStream_t stream) {
  const float* z      = (const float*)d_in[0];
  const float* W1     = (const float*)d_in[1];
  const float* b1     = (const float*)d_in[2];
  const float* g1     = (const float*)d_in[3];
  const float* be1    = (const float*)d_in[4];
  const float* W2     = (const float*)d_in[5];
  const float* b2     = (const float*)d_in[6];
  const float* g2     = (const float*)d_in[7];
  const float* be2    = (const float*)d_in[8];
  const float* W3     = (const float*)d_in[9];
  const float* b3     = (const float*)d_in[10];
  const float* wscale = (const float*)d_in[11];
  const float* initp  = (const float*)d_in[12];
  const int*   nstep  = (const int*)d_in[13];
  float* out = (float*)d_out;

  float* h1     = (float*)d_ws;            // 128*256
  float* h2     = h1 + 128 * 256;          // 128*256
  float* params = h2 + 128 * 256;          // 128*2616

  mlp_ln_relu<<<128, 256, 0, stream>>>(z,  W1, b1, g1, be1, h1, 64);
  mlp_ln_relu<<<128, 256, 0, stream>>>(h1, W2, b2, g2, be2, h2, 256);
  mlp_out<<<128, 256, 0, stream>>>(h2, W3, b3, wscale, params);
  ca_kernel<<<128, 256, SMEM_BYTES, stream>>>(params, initp, nstep, out);
}
